// RelationalReasoning_81003083202647
// MI455X (gfx1250) — matrix operand order repaired and checked by reading, not compiled, not run on hardware
//
#include <hip/hip_runtime.h>

// RelationalReasoning GNN for MI455X (gfx1250, wave32).
// bf16 activations/weights, fp32 accumulate via v_wmma_f32_16x16x32_bf16.
// Weights staged into LDS per workgroup via the Tensor Data Mover (TDM) with
// padded rows; 8 waves/block, grid-stride tile loops; fp32 atomic scatter.

#define N_NODES 100000
#define N_EDGES 1600000
#define HID 64
#define WSTRIDE 72            // LDS row stride in bf16 elems (64 data + 8 pad)
#define NWAVES 8
#define BLK (NWAVES * 32)
#define NT_NODE (N_NODES / 16)   // 6250
#define NT_EDGE (N_EDGES / 16)   // 100000
#define GRID_EDGE 1250           // 1250*8 waves * 10 iters = 100000 tiles exactly
#define GRID_NODE 157            // 1256 wave slots, ~5 iters (guarded)

typedef __bf16 bf16;
typedef __attribute__((ext_vector_type(16))) __bf16 v16bf;
typedef __attribute__((ext_vector_type(8)))  __bf16 v8bf;
typedef __attribute__((ext_vector_type(8)))  float   v8f;
typedef __attribute__((ext_vector_type(4)))  unsigned int v4u;
typedef __attribute__((ext_vector_type(8)))  int v8i;
typedef __attribute__((ext_vector_type(4)))  int v4i;

static __device__ __forceinline__ bf16 f2bf(float f) { return (bf16)f; }

// ------------------------- TDM weight staging -------------------------------
// Copy `rows` x 64-bf16 row-major global tile into LDS with WSTRIDE-padded rows.
// D# layout per CDNA5 ISA: 1-D tensor of `rows*32`
// dwords, pad_enable with 4 dwords of pad after every 32 dwords (=> 72-elem rows).
#if defined(__has_builtin)
#if __has_builtin(__builtin_amdgcn_tensor_load_to_lds)
#define HAVE_TDM 1
#endif
#endif
#ifndef HAVE_TDM
#define HAVE_TDM 0
#endif

#if HAVE_TDM
static __device__ __forceinline__ void tdm_stage_rows(const bf16* g, bf16* lds, int rows) {
  unsigned long long ga = (unsigned long long)(uintptr_t)g;
  unsigned int la = (unsigned int)(uintptr_t)lds;   // low 32 bits = LDS offset
  const int dw = rows * (HID * 2 / 4);              // total dwords (<= 6144)
  v4u g0;
  g0[0] = 1u;                                        // count=1, user descriptor
  g0[1] = la;                                        // lds_addr
  g0[2] = (unsigned int)ga;                          // global_addr[31:0]
  g0[3] = (unsigned int)((ga >> 32) & 0x1FFFFFFull)  // global_addr[56:32]
        | (2u << 30);                                // type=2 ("image")
  v8i g1;
  g1[0] = (2 << 16)        // data_size = 4B
        | (1 << 20)        // pad_enable
        | (4 << 22)        // pad_interval: 2^(4+1)=32 dwords
        | (3 << 25);       // pad_amount: 4 dwords (8 bf16)
  g1[1] = (dw & 0xFFFF) << 16;                       // tensor_dim0[15:0] @ bits 63:48
  g1[2] = ((dw >> 16) & 0xFFFF) | (1 << 16);         // tensor_dim0[31:16], tensor_dim1=1
  g1[3] = (dw & 0xFFFF) << 16;                       // tile_dim0 @ bits 127:112
  g1[4] = 1;                                         // tile_dim1 = 1
  g1[5] = dw;                                        // tensor_dim0_stride[31:0]
  g1[6] = 0;
  g1[7] = 0;
  v4i z4 = {0, 0, 0, 0};
#if defined(__clang_major__) && (__clang_major__ >= 23)
  v8i z8 = {0, 0, 0, 0, 0, 0, 0, 0};
  __builtin_amdgcn_tensor_load_to_lds(g0, g1, z4, z4, z8, 0);
#else
  __builtin_amdgcn_tensor_load_to_lds(g0, g1, z4, z4, 0);
#endif
}
#endif

static __device__ __forceinline__ void stage_weights(const bf16* g, bf16* lds, int rows) {
#if HAVE_TDM
  if (threadIdx.x < 32) tdm_stage_rows(g, lds, rows);   // one TDM issue (wave 0)
#else
  for (int i = threadIdx.x; i < rows * HID; i += BLK)
    lds[(i >> 6) * WSTRIDE + (i & 63)] = g[i];
#endif
}

static __device__ __forceinline__ void stage_commit() {
#if HAVE_TDM
  if (threadIdx.x < 32) __builtin_amdgcn_s_wait_tensorcnt(0);
#endif
  __syncthreads();
}

// ------------------------- WMMA fragment helpers ----------------------------
// Layouts per CDNA5 ISA 7.12.2 (wave32).

// A fragment (16x32 bf16): lane L row M=L&15, kbase=8*(L>>4);
// elems 0..7 -> K+kbase+0..7, elems 8..15 -> K+kbase+16..23.
static __device__ __forceinline__ v16bf a_frag_bf16(const bf16* p, int lane) {
  const int kbase = (lane >> 4) << 3;
  v8bf lo = *(const v8bf*)(p + kbase);
  v8bf hi = *(const v8bf*)(p + kbase + 16);
  v16bf a;
#pragma unroll
  for (int i = 0; i < 8; ++i) { a[i] = lo[i]; a[i + 8] = hi[i]; }
  return a;
}

static __device__ __forceinline__ v16bf a_frag_f32(const float* p, int lane) {
  const int kbase = (lane >> 4) << 3;
  v16bf a;
#pragma unroll
  for (int i = 0; i < 8; ++i) {
    a[i]     = f2bf(p[kbase + i]);
    a[i + 8] = f2bf(p[kbase + 16 + i]);
  }
  return a;
}

// B fragment (32x16 bf16) from LDS-staged weights (row stride WSTRIDE):
// lane L holds column n0+(L&15), kbase=8*(L>>4); elems 0..7 -> K-rows k0+kbase+0..7, elems 8..15 -> k0+kbase+16..23.
static __device__ __forceinline__ v16bf b_frag_lds(const bf16* lds, int k0, int n0, int lane) {
  const int kbase = (lane >> 4) << 3;
  const bf16* p = lds + (k0 + kbase) * WSTRIDE + n0 + (lane & 15);
  v16bf b;
#pragma unroll
  for (int i = 0; i < 8; ++i) { b[i] = p[i * WSTRIDE]; b[i + 8] = p[(16 + i) * WSTRIDE]; }
  return b;
}

#define WMMA_BF16(a, b, c) \
  __builtin_amdgcn_wmma_f32_16x16x32_bf16(false, (a), false, (b), (short)0, (c), false, false)

// ------------------------- utility kernels ----------------------------------

__global__ void cvt_bf16_kernel(const float* __restrict__ s, bf16* __restrict__ d, int n) {
  int i = blockIdx.x * blockDim.x + threadIdx.x;
  if (i < n) d[i] = f2bf(s[i]);
}

__global__ void zero_f32_kernel(float* __restrict__ p, int n) {
  int i = blockIdx.x * blockDim.x + threadIdx.x;
  if (i < n) p[i] = 0.f;
}

// ---- x = node_features @ Wn + bn  (K=128) -> fp32 x, bf16 xh ----
__global__ __launch_bounds__(BLK)
void node_embed_kernel(const float* __restrict__ nf, const bf16* __restrict__ WnH,
                       const float* __restrict__ bn,
                       float* __restrict__ x, bf16* __restrict__ xh) {
  __shared__ bf16 ldsW[128 * WSTRIDE];
  stage_weights(WnH, ldsW, 128);
  stage_commit();
  const int lane = threadIdx.x & 31;
  const int wid  = threadIdx.x >> 5;
  const int nc   = lane & 15;
  float bnv[4];
#pragma unroll
  for (int n = 0; n < 4; ++n) bnv[n] = bn[n * 16 + nc];
  for (int tile = blockIdx.x * NWAVES + wid; tile < NT_NODE; tile += GRID_NODE * NWAVES) {
    const float* rp = nf + (size_t)(tile * 16 + (lane & 15)) * 128;
    v16bf a[4];
#pragma unroll
    for (int kk = 0; kk < 4; ++kk) a[kk] = a_frag_f32(rp + kk * 32, lane);
    v8f acc[4] = {};
#pragma unroll
    for (int kk = 0; kk < 4; ++kk)
#pragma unroll
      for (int n = 0; n < 4; ++n)
        acc[n] = WMMA_BF16(a[kk], b_frag_lds(ldsW, kk * 32, n * 16, lane), acc[n]);
    const int mbase = tile * 16 + ((lane >> 4) << 3);
#pragma unroll
    for (int n = 0; n < 4; ++n)
#pragma unroll
      for (int r = 0; r < 8; ++r) {
        float v = acc[n][r] + bnv[n];
        size_t o = (size_t)(mbase + r) * HID + n * 16 + nc;
        x[o]  = v;
        xh[o] = f2bf(v);
      }
  }
}

// ---- e = edge_features @ We + be  (K=64) -> bf16 eh ----
__global__ __launch_bounds__(BLK)
void edge_embed_kernel(const float* __restrict__ ef, const bf16* __restrict__ WeH,
                       const float* __restrict__ be, bf16* __restrict__ eh) {
  __shared__ bf16 ldsW[64 * WSTRIDE];
  stage_weights(WeH, ldsW, 64);
  stage_commit();
  const int lane = threadIdx.x & 31;
  const int wid  = threadIdx.x >> 5;
  const int nc   = lane & 15;
  float bev[4];
#pragma unroll
  for (int n = 0; n < 4; ++n) bev[n] = be[n * 16 + nc];
  for (int tile = blockIdx.x * NWAVES + wid; tile < NT_EDGE; tile += GRID_EDGE * NWAVES) {
    const float* rp = ef + (size_t)(tile * 16 + (lane & 15)) * 64;
    v16bf a[2];
#pragma unroll
    for (int kk = 0; kk < 2; ++kk) a[kk] = a_frag_f32(rp + kk * 32, lane);
    v8f acc[4] = {};
#pragma unroll
    for (int kk = 0; kk < 2; ++kk)
#pragma unroll
      for (int n = 0; n < 4; ++n)
        acc[n] = WMMA_BF16(a[kk], b_frag_lds(ldsW, kk * 32, n * 16, lane), acc[n]);
    const int mbase = tile * 16 + ((lane >> 4) << 3);
#pragma unroll
    for (int n = 0; n < 4; ++n)
#pragma unroll
      for (int r = 0; r < 8; ++r)
        eh[(size_t)(mbase + r) * HID + n * 16 + nc] = f2bf(acc[n][r] + bev[n]);
  }
}

// ---- edge message MLP + fp32 atomic scatter:
// mi=[xh[src]|xh[dst]|eh[e]] (K=192); m=relu(mi@W1+b1)@W2+b2; agg[dst]+=m ----
__global__ __launch_bounds__(BLK)
void edge_mlp_kernel(const bf16* __restrict__ xh, const bf16* __restrict__ eh,
                     const int* __restrict__ ei,
                     const bf16* __restrict__ W1H, const float* __restrict__ b1,
                     const bf16* __restrict__ W2H, const float* __restrict__ b2,
                     float* __restrict__ agg) {
  __shared__ bf16 ldsW1[192 * WSTRIDE];
  __shared__ bf16 ldsW2[64 * WSTRIDE];
  __shared__ bf16 hbuf[NWAVES][16 * WSTRIDE];
  __shared__ int  dbuf[NWAVES][16];
  stage_weights(W1H, ldsW1, 192);
  stage_weights(W2H, ldsW2, 64);   // second TDM issue, same wave: in-order
  stage_commit();

  const int lane  = threadIdx.x & 31;
  const int wid   = threadIdx.x >> 5;
  const int erow  = lane & 15;
  const int nc    = lane & 15;
  const int rbase = (lane >> 4) << 3;
  float b1v[4], b2v[4];
#pragma unroll
  for (int n = 0; n < 4; ++n) { b1v[n] = b1[n * 16 + nc]; b2v[n] = b2[n * 16 + nc]; }
  bf16* hb = hbuf[wid];
  int*  db = dbuf[wid];

  for (int tile = blockIdx.x * NWAVES + wid; tile < NT_EDGE; tile += GRID_EDGE * NWAVES) {
    const int eid = tile * 16 + erow;
    const int src = ei[eid];
    const int dst = ei[N_EDGES + eid];
    if (lane < 16) db[lane] = dst;

    // all six K-step A fragments up front (overlapped gather latency)
    const bf16* bs = xh + (size_t)src * HID;
    const bf16* bd = xh + (size_t)dst * HID;
    const bf16* bg = eh + (size_t)eid * HID;
    v16bf a[6];
    a[0] = a_frag_bf16(bs,      lane);  a[1] = a_frag_bf16(bs + 32, lane);
    a[2] = a_frag_bf16(bd,      lane);  a[3] = a_frag_bf16(bd + 32, lane);
    a[4] = a_frag_bf16(bg,      lane);  a[5] = a_frag_bf16(bg + 32, lane);

    v8f acc[4] = {};
#pragma unroll
    for (int kk = 0; kk < 6; ++kk)
#pragma unroll
      for (int n = 0; n < 4; ++n)
        acc[n] = WMMA_BF16(a[kk], b_frag_lds(ldsW1, kk * 32, n * 16, lane), acc[n]);

    // relu(h+b1) -> per-wave LDS tile (same-wave DS ordering; no block barrier)
#pragma unroll
    for (int n = 0; n < 4; ++n)
#pragma unroll
      for (int r = 0; r < 8; ++r) {
        float v = acc[n][r] + b1v[n];
        hb[(rbase + r) * WSTRIDE + n * 16 + nc] = f2bf(v > 0.f ? v : 0.f);
      }

    v8f acc2[4] = {};
#pragma unroll
    for (int k0 = 0; k0 < 64; k0 += 32) {
      v16bf ah = a_frag_bf16(hb + erow * WSTRIDE + k0, lane);
#pragma unroll
      for (int n = 0; n < 4; ++n)
        acc2[n] = WMMA_BF16(ah, b_frag_lds(ldsW2, k0, n * 16, lane), acc2[n]);
    }

    // segment-sum scatter (fp32 atomics)
#pragma unroll
    for (int n = 0; n < 4; ++n)
#pragma unroll
      for (int r = 0; r < 8; ++r) {
        float v = acc2[n][r] + b2v[n];
        int d = db[rbase + r];
        __hip_atomic_fetch_add(&agg[(size_t)d * HID + n * 16 + nc], v,
                               __ATOMIC_RELAXED, __HIP_MEMORY_SCOPE_AGENT);
      }
  }
}

// ---- node update: x' = relu([x|agg]@U1+bu1)@U2 + bu2 + x  (K=128) ----
__global__ __launch_bounds__(BLK)
void node_update_kernel(const float* __restrict__ x, const bf16* __restrict__ xh,
                        const float* __restrict__ agg,
                        const bf16* __restrict__ U1H, const float* __restrict__ bu1,
                        const bf16* __restrict__ U2H, const float* __restrict__ bu2,
                        float* __restrict__ xo, bf16* __restrict__ xoh) {
  __shared__ bf16 ldsU1[128 * WSTRIDE];
  __shared__ bf16 ldsU2[64 * WSTRIDE];
  __shared__ bf16 hbuf[NWAVES][16 * WSTRIDE];
  stage_weights(U1H, ldsU1, 128);
  stage_weights(U2H, ldsU2, 64);
  stage_commit();

  const int lane  = threadIdx.x & 31;
  const int wid   = threadIdx.x >> 5;
  const int nrow  = lane & 15;
  const int nc    = lane & 15;
  const int rbase = (lane >> 4) << 3;
  float bu1v[4], bu2v[4];
#pragma unroll
  for (int n = 0; n < 4; ++n) { bu1v[n] = bu1[n * 16 + nc]; bu2v[n] = bu2[n * 16 + nc]; }
  bf16* hb = hbuf[wid];

  for (int tile = blockIdx.x * NWAVES + wid; tile < NT_NODE; tile += GRID_NODE * NWAVES) {
    const int node = tile * 16 + nrow;
    v16bf a[4];
    a[0] = a_frag_bf16(xh  + (size_t)node * HID,      lane);
    a[1] = a_frag_bf16(xh  + (size_t)node * HID + 32, lane);
    a[2] = a_frag_f32 (agg + (size_t)node * HID,      lane);
    a[3] = a_frag_f32 (agg + (size_t)node * HID + 32, lane);

    v8f acc[4] = {};
#pragma unroll
    for (int kk = 0; kk < 4; ++kk)
#pragma unroll
      for (int n = 0; n < 4; ++n)
        acc[n] = WMMA_BF16(a[kk], b_frag_lds(ldsU1, kk * 32, n * 16, lane), acc[n]);

#pragma unroll
    for (int n = 0; n < 4; ++n)
#pragma unroll
      for (int r = 0; r < 8; ++r) {
        float v = acc[n][r] + bu1v[n];
        hb[(rbase + r) * WSTRIDE + n * 16 + nc] = f2bf(v > 0.f ? v : 0.f);
      }

    v8f acc2[4] = {};
#pragma unroll
    for (int k0 = 0; k0 < 64; k0 += 32) {
      v16bf ah = a_frag_bf16(hb + nrow * WSTRIDE + k0, lane);
#pragma unroll
      for (int n = 0; n < 4; ++n)
        acc2[n] = WMMA_BF16(ah, b_frag_lds(ldsU2, k0, n * 16, lane), acc2[n]);
    }

    const int mbase = tile * 16 + rbase;
#pragma unroll
    for (int n = 0; n < 4; ++n)
#pragma unroll
      for (int r = 0; r < 8; ++r) {
        size_t o = (size_t)(mbase + r) * HID + n * 16 + nc;
        float v = acc2[n][r] + bu2v[n] + x[o];   // residual
        xo[o] = v;
        if (xoh) xoh[o] = f2bf(v);
      }
  }
}

// ------------------------- host orchestration -------------------------------
extern "C" void kernel_launch(void* const* d_in, const int* in_sizes, int n_in,
                              void* d_out, int out_size, void* d_ws, size_t ws_size,
                              hipStream_t stream) {
  (void)in_sizes; (void)n_in; (void)out_size; (void)ws_size;
  const float* nf  = (const float*)d_in[0];
  const float* ef  = (const float*)d_in[1];
  const int*   ei  = (const int*)  d_in[2];
  const float* Wn  = (const float*)d_in[3];  const float* bn  = (const float*)d_in[4];
  const float* We  = (const float*)d_in[5];  const float* be  = (const float*)d_in[6];
  const float* W1  = (const float*)d_in[7];  const float* b1  = (const float*)d_in[8];
  const float* W2  = (const float*)d_in[9];  const float* b2  = (const float*)d_in[10];
  const float* U1  = (const float*)d_in[11]; const float* bu1 = (const float*)d_in[12];
  const float* U2  = (const float*)d_in[13]; const float* bu2 = (const float*)d_in[14];

  uintptr_t base = (uintptr_t)d_ws;
  size_t off = 0;
  auto carve = [&](size_t bytes) -> void* {
    void* p = (void*)(base + off);
    off += (bytes + 255) & ~(size_t)255;
    return p;
  };
  bf16* WnH = (bf16*)carve(128 * 64 * 2);
  bf16* WeH = (bf16*)carve( 64 * 64 * 2);
  bf16* W1H = (bf16*)carve(192 * 64 * 2);
  bf16* W2H = (bf16*)carve( 64 * 64 * 2);
  bf16* U1H = (bf16*)carve(128 * 64 * 2);
  bf16* U2H = (bf16*)carve( 64 * 64 * 2);
  float* xA  = (float*)carve((size_t)N_NODES * HID * 4);
  float* xB  = (float*)carve((size_t)N_NODES * HID * 4);
  bf16*  xhA = (bf16*) carve((size_t)N_NODES * HID * 2);
  bf16*  xhB = (bf16*) carve((size_t)N_NODES * HID * 2);
  float* agg = (float*)carve((size_t)N_NODES * HID * 4);
  bf16*  eh  = (bf16*) carve((size_t)N_EDGES * HID * 2);

  cvt_bf16_kernel<<<(128 * 64 + 255) / 256, 256, 0, stream>>>(Wn, WnH, 128 * 64);
  cvt_bf16_kernel<<<( 64 * 64 + 255) / 256, 256, 0, stream>>>(We, WeH,  64 * 64);
  cvt_bf16_kernel<<<(192 * 64 + 255) / 256, 256, 0, stream>>>(W1, W1H, 192 * 64);
  cvt_bf16_kernel<<<( 64 * 64 + 255) / 256, 256, 0, stream>>>(W2, W2H,  64 * 64);
  cvt_bf16_kernel<<<(128 * 64 + 255) / 256, 256, 0, stream>>>(U1, U1H, 128 * 64);
  cvt_bf16_kernel<<<( 64 * 64 + 255) / 256, 256, 0, stream>>>(U2, U2H,  64 * 64);

  node_embed_kernel<<<GRID_NODE, BLK, 0, stream>>>(nf, WnH, bn, xA, xhA);
  edge_embed_kernel<<<GRID_EDGE, BLK, 0, stream>>>(ef, WeH, be, eh);

  float* xc = xA;  bf16* xhc = xhA;
  float* xn = xB;  bf16* xhn = xhB;
  for (int layer = 0; layer < 3; ++layer) {
    zero_f32_kernel<<<(N_NODES * HID + 255) / 256, 256, 0, stream>>>(agg, N_NODES * HID);
    edge_mlp_kernel<<<GRID_EDGE, BLK, 0, stream>>>(xhc, eh, ei, W1H, b1, W2H, b2, agg);
    float* outx = (layer == 2) ? (float*)d_out : xn;
    bf16*  outh = (layer == 2) ? nullptr : xhn;
    node_update_kernel<<<GRID_NODE, BLK, 0, stream>>>(xc, xhc, agg, U1H, bu1, U2H, bu2, outx, outh);
    float* tx = xc; xc = xn; xn = tx;
    bf16*  th = xhc; xhc = xhn; xhn = th;
  }
}
